// Attention_18511309046424
// MI455X (gfx1250) — matrix operand order repaired and checked by reading, not compiled, not run on hardware
//
#include <hip/hip_runtime.h>
#include <stdint.h>

#define BATCH   4
#define NSEQ    2048
#define DMODEL  1024
#define HEADS   16
#define DH      64
#define SCALE_F 0.125f   // 64^-0.5

typedef __attribute__((ext_vector_type(16))) __bf16 v16bf;
typedef __attribute__((ext_vector_type(8)))  float  v8f;
typedef __attribute__((ext_vector_type(4)))  unsigned int v4u;
typedef __attribute__((ext_vector_type(8)))  int v8i;
typedef __attribute__((ext_vector_type(4)))  int v4i;

#if defined(__HIP_DEVICE_COMPILE__) && __has_builtin(__builtin_amdgcn_tensor_load_to_lds)
#define HAS_TDM 1
#else
#define HAS_TDM 0
#endif

// ---------- fp32 -> bf16, round-to-nearest-even ----------
__device__ __forceinline__ unsigned short f32_to_bf16(float f) {
    unsigned int u = __float_as_uint(f);
    u += 0x7FFFu + ((u >> 16) & 1u);
    return (unsigned short)(u >> 16);
}

// ---------- WMMA wrapper ----------
__device__ __forceinline__ v8f wmma_bf16(v16bf a, v16bf b, v8f c) {
    return __builtin_amdgcn_wmma_f32_16x16x32_bf16(
        false, a, false, b, (short)0, c, false, false);
}

// ---------- A fragment: 16x32 bf16 (MxK), src row-major [*, ld] ----------
__device__ __forceinline__ v16bf load_a_frag(const unsigned short* __restrict__ src,
                                             int ld, int m_base, int k0) {
    const int lane = threadIdx.x & 31;
    const int m = m_base + (lane & 15);
    const int h = lane >> 4;
    union { v16bf v; unsigned int u[8]; } f;
    const unsigned short* p = src + (size_t)m * ld + k0 + 16 * h;
#pragma unroll
    for (int j = 0; j < 4; ++j) f.u[j] = *(const unsigned int*)(p + 2 * j);
#pragma unroll
    for (int j = 0; j < 4; ++j) f.u[4 + j] = *(const unsigned int*)(p + 8 + 2 * j);
    return f.v;
}

// ---------- B fragment (32x16, KxN) from contiguous 16-element run (LDS or global) ----------
// caller passes p = base + n_row*ld + k0 + 16*h ; elements e: K = 16h + e
__device__ __forceinline__ v16bf load_b_frag_at(const unsigned short* p) {
    union { v16bf v; unsigned int u[8]; } f;
#pragma unroll
    for (int j = 0; j < 8; ++j) f.u[j] = *(const unsigned int*)(p + 2 * j);
    return f.v;
}

#if HAS_TDM
// ---------- TDM 2D tile load: global (row-major, stride0 elems of 2B) -> LDS ----------
// D# packing per CDNA5 ISA (data_size=2 bytes, count=1, type=2)
__device__ __forceinline__ void tdm_load_2d(const void* gsrc, unsigned lds_off,
                                            unsigned td0, unsigned td1,
                                            unsigned tile0, unsigned tile1,
                                            unsigned stride0) {
    unsigned long long ga = (unsigned long long)(uintptr_t)gsrc;
    v4u g0;
    g0[0] = 1u;                                   // count=1, is_restore=0, gather off
    g0[1] = lds_off;                              // lds_addr (bytes)
    g0[2] = (unsigned)ga;                         // global_addr[31:0]
    g0[3] = ((unsigned)(ga >> 32) & 0x01FFFFFFu) | 0x80000000u;  // addr[56:32] | type=2
    v8i g1;
    g1[0] = (int)(1u << 16);                      // data_size = 1 (2 bytes)
    g1[1] = (int)(td0 << 16);                     // tensor_dim0[15:0]
    g1[2] = (int)((td0 >> 16) | (td1 << 16));     // tensor_dim0[31:16] | tensor_dim1[15:0]
    g1[3] = (int)((td1 >> 16) | (tile0 << 16));   // tensor_dim1[31:16] | tile_dim0
    g1[4] = (int)(tile1 & 0xFFFFu);               // tile_dim1 | tile_dim2=0
    g1[5] = (int)stride0;                         // tensor_dim0_stride[31:0]
    g1[6] = 0;                                    // stride0[47:32] | stride1[15:0]
    g1[7] = 0;
    v4i z = {0, 0, 0, 0};
#if defined(__clang_major__) && __clang_major__ >= 23
    v8i z8 = {0, 0, 0, 0, 0, 0, 0, 0};
    __builtin_amdgcn_tensor_load_to_lds(g0, g1, z, z, z8, 0);
#else
    __builtin_amdgcn_tensor_load_to_lds(g0, g1, z, z, 0);
#endif
}
#endif

// ---------- elementwise fp32 -> bf16 ----------
__global__ void cvt_f32_to_bf16(const float* __restrict__ in,
                                unsigned short* __restrict__ out, int nelem) {
    int i = blockIdx.x * blockDim.x + threadIdx.x;
    if (i < nelem) out[i] = f32_to_bf16(in[i]);
}

// ---------- GEMM: C[M x 1024] = A[M x 1024] * W^T ----------
// block = 256 threads (8 waves); tile 256(M) x 64(N); wave: 32 rows x 64 cols (8 accums)
// W tile [64 x 32k] staged to LDS via TDM (double buffered); B-fragments read from LDS.
// mode 0: bf16 -> [b,h,n,dh]  | mode 1: bf16 -> [b,h,dh,n] | mode 2: fp32 + bias
__global__ __launch_bounds__(256) void gemm_nt_bf16(
    const unsigned short* __restrict__ A, const unsigned short* __restrict__ W,
    unsigned short* __restrict__ outb, float* __restrict__ outf,
    const float* __restrict__ bias, int Kd, int mode)
{
    __shared__ unsigned short ldsW[2][64 * 32];   // 8 KB
    const int tid  = threadIdx.x;
    const int lane = tid & 31;
    const int wave = tid >> 5;
    const int h = lane >> 4;
    const int n = lane & 15;
    const int m0 = blockIdx.x * 256 + wave * 32;
    const int n0 = blockIdx.y * 64;

    v8f c0[4] = {}, c1[4] = {};
    const int nsteps = Kd / 32;

#if HAS_TDM
    if (tid < 32)
        tdm_load_2d(W + (size_t)n0 * Kd, (unsigned)(uintptr_t)&ldsW[0][0],
                    (unsigned)Kd, (unsigned)DMODEL, 32u, 64u, (unsigned)Kd);
#endif
    for (int s = 0; s < nsteps; ++s) {
        const int k0 = s * 32;
#if HAS_TDM
        if (tid < 32) {
            if (s + 1 < nsteps) {
                tdm_load_2d(W + (size_t)n0 * Kd + (k0 + 32),
                            (unsigned)(uintptr_t)&ldsW[(s + 1) & 1][0],
                            (unsigned)Kd, (unsigned)DMODEL, 32u, 64u, (unsigned)Kd);
                __builtin_amdgcn_s_wait_tensorcnt(1);
            } else {
                __builtin_amdgcn_s_wait_tensorcnt(0);
            }
        }
        __syncthreads();
        const unsigned short* bw = &ldsW[s & 1][0];
#else
        __syncthreads();
        {   // cooperative copy: 256 threads x 16B = 4 KB tile [64 rows x 32 k]
            const int row = tid >> 2, seg = tid & 3;
            *(uint4*)&ldsW[0][row * 32 + seg * 8] =
                *(const uint4*)&W[(size_t)(n0 + row) * Kd + k0 + seg * 8];
        }
        __syncthreads();
        const unsigned short* bw = &ldsW[0][0];
#endif
        v16bf a0 = load_a_frag(A, Kd, m0,      k0);
        v16bf a1 = load_a_frag(A, Kd, m0 + 16, k0);
#pragma unroll
        for (int t = 0; t < 4; ++t) {
            v16bf b = load_b_frag_at(bw + (16 * t + n) * 32 + 16 * h);
            c0[t] = wmma_bf16(a0, b, c0[t]);
            c1[t] = wmma_bf16(a1, b, c1[t]);
        }
        __syncthreads();
    }

#pragma unroll
    for (int i = 0; i < 2; ++i) {
#pragma unroll
        for (int t = 0; t < 4; ++t) {
            v8f c = i ? c1[t] : c0[t];
            const int col = n0 + 16 * t + n;
#pragma unroll
            for (int r = 0; r < 8; ++r) {
                const int row = m0 + 16 * i + r + 8 * h;
                const float v = c[r];
                if (mode == 2) {
                    outf[(size_t)row * DMODEL + col] = v + bias[col];
                } else {
                    const int b = row >> 11, ns = row & (NSEQ - 1);
                    const int head = col >> 6, dhi = col & (DH - 1);
                    size_t idx;
                    if (mode == 0)
                        idx = (((size_t)(b * HEADS + head) * NSEQ) + ns) * DH + dhi;
                    else
                        idx = (((size_t)(b * HEADS + head) * DH) + dhi) * NSEQ + ns;
                    outb[idx] = f32_to_bf16(v);
                }
            }
        }
    }
}

// ---------- Flash attention ----------
// block = 256 threads (8 waves) per 128-query tile of one (b,h).
// K tile [32 keys x 64 dh] and V^T tile [64 dh x 32 keys] staged via TDM, double buffered,
// shared by all 8 waves. Online softmax in fp32; P re-laid out D->A through per-wave LDS.
__global__ __launch_bounds__(256) void flash_attn(
    const unsigned short* __restrict__ Q, const unsigned short* __restrict__ K,
    const unsigned short* __restrict__ VT, unsigned short* __restrict__ Ob)
{
    __shared__ unsigned short ldsK[2][32 * 64];   // [key][dh]   8 KB
    __shared__ unsigned short ldsV[2][64 * 32];   // [dh][key]   8 KB
    __shared__ unsigned short ldsP[8][16 * 32];   // per-wave P  8 KB

    const int tid  = threadIdx.x;
    const int lane = tid & 31;
    const int wave = tid >> 5;
    const int h = lane >> 4;
    const int n = lane & 15;
    const int bh = blockIdx.y;
    const int q0 = blockIdx.x * 128 + wave * 16;

    const unsigned short* Qp = Q  + (size_t)bh * NSEQ * DH;
    const unsigned short* Kp = K  + (size_t)bh * NSEQ * DH;
    const unsigned short* Vp = VT + (size_t)bh * DH * NSEQ;

    const v16bf aq0 = load_a_frag(Qp, DH, q0, 0);
    const v16bf aq1 = load_a_frag(Qp, DH, q0, 32);

    float mrow[8], lrow[8];
#pragma unroll
    for (int r = 0; r < 8; ++r) { mrow[r] = -3.0e38f; lrow[r] = 0.0f; }
    v8f o0 = {}, o1 = {}, o2 = {}, o3 = {};

    unsigned short* pw = &ldsP[wave][0];
    const int nsteps = NSEQ / 32;

#if HAS_TDM
    if (tid < 32) {
        tdm_load_2d(Kp, (unsigned)(uintptr_t)&ldsK[0][0], DH, NSEQ, DH, 32u, DH);
        tdm_load_2d(Vp, (unsigned)(uintptr_t)&ldsV[0][0], NSEQ, DH, 32u, DH, NSEQ);
    }
#endif
    for (int s = 0; s < nsteps; ++s) {
        const int j0 = s * 32;
#if HAS_TDM
        if (tid < 32) {
            if (s + 1 < nsteps) {
                const int jn = j0 + 32;
                tdm_load_2d(Kp + (size_t)jn * DH,
                            (unsigned)(uintptr_t)&ldsK[(s + 1) & 1][0], DH, NSEQ, DH, 32u, DH);
                tdm_load_2d(Vp + jn,
                            (unsigned)(uintptr_t)&ldsV[(s + 1) & 1][0], NSEQ, DH, 32u, DH, NSEQ);
                __builtin_amdgcn_s_wait_tensorcnt(2);
            } else {
                __builtin_amdgcn_s_wait_tensorcnt(0);
            }
        }
        __syncthreads();
        const unsigned short* kb = &ldsK[s & 1][0];
        const unsigned short* vb = &ldsV[s & 1][0];
#else
        __syncthreads();
        {   // cooperative copy of K tile (32x64) and V tile (64x32), 16B per thread each
            const int kr = tid >> 3, ks = tid & 7;
            *(uint4*)&ldsK[0][kr * 64 + ks * 8] =
                *(const uint4*)&Kp[(size_t)(j0 + kr) * DH + ks * 8];
            const int vr = tid >> 2, vs = tid & 3;
            *(uint4*)&ldsV[0][vr * 32 + vs * 8] =
                *(const uint4*)&Vp[(size_t)vr * NSEQ + j0 + vs * 8];
        }
        __syncthreads();
        const unsigned short* kb = &ldsK[0][0];
        const unsigned short* vb = &ldsV[0][0];
#endif
        // S = Q K^T : two 16x16 fp32 tiles over 32 keys
        v8f s0 = {}, s1 = {};
        s0 = wmma_bf16(aq0, load_b_frag_at(kb + n * 64 + 16 * h), s0);
        s0 = wmma_bf16(aq1, load_b_frag_at(kb + n * 64 + 32 + 16 * h), s0);
        s1 = wmma_bf16(aq0, load_b_frag_at(kb + (16 + n) * 64 + 16 * h), s1);
        s1 = wmma_bf16(aq1, load_b_frag_at(kb + (16 + n) * 64 + 32 + 16 * h), s1);

        float corr[8];
#pragma unroll
        for (int r = 0; r < 8; ++r) {
            float v0 = s0[r] * SCALE_F;
            float v1 = s1[r] * SCALE_F;
            float mx = fmaxf(v0, v1);
            mx = fmaxf(mx, __shfl_xor(mx, 1, 32));
            mx = fmaxf(mx, __shfl_xor(mx, 2, 32));
            mx = fmaxf(mx, __shfl_xor(mx, 4, 32));
            mx = fmaxf(mx, __shfl_xor(mx, 8, 32));
            const float mnew = fmaxf(mrow[r], mx);
            const float e0 = __expf(v0 - mnew);
            const float e1 = __expf(v1 - mnew);
            corr[r] = __expf(mrow[r] - mnew);
            mrow[r] = mnew;
            float rs = e0 + e1;
            rs += __shfl_xor(rs, 1, 32);
            rs += __shfl_xor(rs, 2, 32);
            rs += __shfl_xor(rs, 4, 32);
            rs += __shfl_xor(rs, 8, 32);
            lrow[r] = lrow[r] * corr[r] + rs;
            pw[(r + 8 * h) * 32 + n]      = f32_to_bf16(e0);
            pw[(r + 8 * h) * 32 + n + 16] = f32_to_bf16(e1);
            o0[r] *= corr[r]; o1[r] *= corr[r]; o2[r] *= corr[r]; o3[r] *= corr[r];
        }

        asm volatile("s_wait_dscnt 0" ::: "memory");   // same-wave LDS RAW drain

        union { v16bf v; unsigned int u[8]; } pf;
        {
            const unsigned short* p = pw + (lane & 15) * 32 + 16 * h;
#pragma unroll
            for (int j = 0; j < 4; ++j) pf.u[j]     = *(const unsigned int*)(p + 2 * j);
#pragma unroll
            for (int j = 0; j < 4; ++j) pf.u[4 + j] = *(const unsigned int*)(p + 8 + 2 * j);
        }

        o0 = wmma_bf16(pf.v, load_b_frag_at(vb + n * 32 + 16 * h), o0);
        o1 = wmma_bf16(pf.v, load_b_frag_at(vb + (16 + n) * 32 + 16 * h), o1);
        o2 = wmma_bf16(pf.v, load_b_frag_at(vb + (32 + n) * 32 + 16 * h), o2);
        o3 = wmma_bf16(pf.v, load_b_frag_at(vb + (48 + n) * 32 + 16 * h), o3);

        __syncthreads();
    }

    const int b = bh >> 4, head = bh & 15;
#pragma unroll
    for (int r = 0; r < 8; ++r) {
        const float inv = 1.0f / lrow[r];
        const int row = q0 + r + 8 * h;
        const size_t base = ((size_t)b * NSEQ + row) * DMODEL + head * DH + n;
        Ob[base +  0] = f32_to_bf16(o0[r] * inv);
        Ob[base + 16] = f32_to_bf16(o1[r] * inv);
        Ob[base + 32] = f32_to_bf16(o2[r] * inv);
        Ob[base + 48] = f32_to_bf16(o3[r] * inv);
    }
}

extern "C" void kernel_launch(void* const* d_in, const int* in_sizes, int n_in,
                              void* d_out, int out_size, void* d_ws, size_t ws_size,
                              hipStream_t stream)
{
    (void)in_sizes; (void)n_in; (void)out_size; (void)ws_size;
    const float* x  = (const float*)d_in[0];
    const float* Wq = (const float*)d_in[1];
    const float* Wk = (const float*)d_in[2];
    const float* Wv = (const float*)d_in[3];
    const float* Wo = (const float*)d_in[4];
    const float* bo = (const float*)d_in[5];
    float* out = (float*)d_out;

    const size_t M  = (size_t)BATCH * NSEQ;        // 8192
    const size_t XE = M * DMODEL;
    const size_t WE = (size_t)DMODEL * DMODEL;

    unsigned short* xb  = (unsigned short*)d_ws;
    unsigned short* wqb = xb  + XE;
    unsigned short* wkb = wqb + WE;
    unsigned short* wvb = wkb + WE;
    unsigned short* wob = wvb + WE;
    unsigned short* Qb  = wob + WE;   // [b,h,n,dh]
    unsigned short* Kb  = Qb  + XE;   // [b,h,n,dh]
    unsigned short* VTb = Kb  + XE;   // [b,h,dh,n]
    unsigned short* AOb = VTb + XE;   // [b,n,h*dh]

    cvt_f32_to_bf16<<<(int)((XE + 255) / 256), 256, 0, stream>>>(x,  xb,  (int)XE);
    cvt_f32_to_bf16<<<(int)((WE + 255) / 256), 256, 0, stream>>>(Wq, wqb, (int)WE);
    cvt_f32_to_bf16<<<(int)((WE + 255) / 256), 256, 0, stream>>>(Wk, wkb, (int)WE);
    cvt_f32_to_bf16<<<(int)((WE + 255) / 256), 256, 0, stream>>>(Wv, wvb, (int)WE);
    cvt_f32_to_bf16<<<(int)((WE + 255) / 256), 256, 0, stream>>>(Wo, wob, (int)WE);

    dim3 gg((unsigned)(M / 256), DMODEL / 64);
    gemm_nt_bf16<<<gg, 256, 0, stream>>>(xb, wqb, Qb,  nullptr, nullptr, DMODEL, 0);
    gemm_nt_bf16<<<gg, 256, 0, stream>>>(xb, wkb, Kb,  nullptr, nullptr, DMODEL, 0);
    gemm_nt_bf16<<<gg, 256, 0, stream>>>(xb, wvb, VTb, nullptr, nullptr, DMODEL, 1);

    dim3 ga(NSEQ / 128, BATCH * HEADS);
    flash_attn<<<ga, 256, 0, stream>>>(Qb, Kb, VTb, AOb);

    gemm_nt_bf16<<<gg, 256, 0, stream>>>(AOb, wob, nullptr, out, bo, DMODEL, 2);
}
